// AttnDecoderLSTM_LongCat_59055800320107
// MI455X (gfx1250) — matrix operand order repaired and checked by reading, not compiled, not run on hardware
//
#include <hip/hip_runtime.h>
#include <hip/hip_bf16.h>
#include <cstdint>

// ---------------- types ----------------
typedef __bf16 bf16_t;
typedef __attribute__((ext_vector_type(16))) __bf16 v16bf;
typedef __attribute__((ext_vector_type(8)))  __bf16 v8bf;
typedef __attribute__((ext_vector_type(8)))  float  v8f;

union FragBF16 { v16bf v; v8bf h[2]; };

// ---------------- constants ----------------
#define BATCH 512
#define SEQ   80
#define NV    36
#define NCAND 16
#define ANGLE 128
#define EMBD  64
#define HIDD  512
#define FEATD 2176
#define GATES 2048   // 4*HID
#define KSTAGE 64    // K elements staged into LDS per double-buffer slot

// ---------------- CDNA5 async global->LDS path (guarded) ----------------
#define USE_ASYNC_LDS 0
#if defined(__has_builtin)
#if __has_builtin(__builtin_amdgcn_global_load_async_to_lds_b128) && \
    __has_builtin(__builtin_amdgcn_s_wait_asynccnt)
#undef USE_ASYNC_LDS
#define USE_ASYNC_LDS 1
#endif
#endif

// builtin signature (from hipcc diagnostic): params are pointers to
// __attribute__((vector_size(16))) int in global (AS1) / shared (AS3) space.
typedef int v4i_gcc __attribute__((vector_size(16)));
typedef __attribute__((address_space(1))) v4i_gcc as1_v4i;
typedef __attribute__((address_space(3))) v4i_gcc as3_v4i;

__device__ __forceinline__ void copy16_to_lds(const bf16_t* g, bf16_t* l) {
#if USE_ASYNC_LDS
    __builtin_amdgcn_global_load_async_to_lds_b128((as1_v4i*)g, (as3_v4i*)l, 0, 0);
#else
    *(v8bf*)l = *(const v8bf*)g;
#endif
}

__device__ __forceinline__ void async_stage_wait() {
#if USE_ASYNC_LDS
    __builtin_amdgcn_s_wait_asynccnt(0);
#endif
}

// ---------------- fp32 -> bf16 convert ----------------
__global__ void cvt_bf16_kernel(const float* __restrict__ src, bf16_t* __restrict__ dst, int n) {
    int i = blockIdx.x * blockDim.x + threadIdx.x;
    int stride = gridDim.x * blockDim.x;
    for (; i < n; i += stride) dst[i] = (bf16_t)src[i];
}

// ---------------- bf16 WMMA GEMM: C[M,N] = op(A[M,K] @ B[N,K]^T + bias (+C)) ----------------
// flags bit0: accumulate into existing C ; bit1: tanh epilogue
// Block = 8 waves; each wave owns a 16x64 strip (4 WMMA accumulators).
// The 16-row A tile is shared by all waves: staged into LDS (double-buffered,
// async global->LDS when available) so the next stage overlaps current WMMAs.
// Requires K % 64 == 0 (holds for all GEMMs here: 64/128/512/2176).
__global__ void gemm_bf16_wmma(const bf16_t* __restrict__ A, int lda,
                               const bf16_t* __restrict__ B, int ldb,
                               float* __restrict__ C, int ldc,
                               const float* __restrict__ bias,
                               int M, int N, int K, int flags) {
    __shared__ bf16_t aLds[2][16 * KSTAGE];      // 2 x 2KB
    const int tid  = threadIdx.x;
    const int lane = tid & 31;
    const int wave = tid >> 5;
    const int col0 = (blockIdx.x * 8 + wave) * 64;
    const bool active = (col0 < N);              // wave-uniform
    const int tileM = blockIdx.y;
    const int hi = lane >> 4;
    const int lo = lane & 15;
    const int rowBase = tileM << 4;

    // Cooperative A staging: threads 0..127 copy 16x64 bf16 (16B each)
    const int srow = tid >> 3;                   // 0..15 when tid<128
    const int sseg = (tid & 7) * 8;              // bf16 element offset 0..56
    const bf16_t* aSrc = A + (size_t)(rowBase + srow) * lda + sseg;
    bf16_t* aDst0 = &aLds[0][srow * KSTAGE + sseg];
    bf16_t* aDst1 = &aLds[1][srow * KSTAGE + sseg];

    v8f acc[4];
#pragma unroll
    for (int t = 0; t < 4; ++t) acc[t] = (v8f){0.f,0.f,0.f,0.f,0.f,0.f,0.f,0.f};

    if (tid < 128) copy16_to_lds(aSrc, aDst0);   // prologue stage 0

    const int nStages = K / KSTAGE;
    for (int s = 0; s < nStages; ++s) {
        const int cur = s & 1;
        if (tid < 128) async_stage_wait();       // our async stage for `cur` done
        __syncthreads();                         // publish stage; all reads of buf cur^1 done
        if ((s + 1 < nStages) && tid < 128)
            copy16_to_lds(aSrc + (size_t)(s + 1) * KSTAGE, cur ? aDst0 : aDst1);

        if (active) {
#pragma unroll
            for (int kk = 0; kk < KSTAGE; kk += 32) {
                // A 16x32 fragment from LDS (lanes0-15: K0-7/16-23, lanes16-31: K8-15/24-31)
                FragBF16 af;
                const bf16_t* ap = &aLds[cur][lo * KSTAGE + kk + hi * 8];
                af.h[0] = *(const v8bf*)(ap);
                af.h[1] = *(const v8bf*)(ap + 16);
                const int kg = s * KSTAGE + kk;
#pragma unroll
                for (int t = 0; t < 4; ++t) {
                    // B 32x16 fragment from global: lanes0-15 K0-7/16-23, lanes16-31 K8-15/24-31
                    const int colB = col0 + t * 16 + lo;
                    const bf16_t* bp = B + (size_t)colB * ldb + kg + hi * 8;
                    FragBF16 bf;
                    bf.h[0] = *(const v8bf*)(bp);
                    bf.h[1] = *(const v8bf*)(bp + 16);
                    acc[t] = __builtin_amdgcn_wmma_f32_16x16x32_bf16(
                        false, af.v, false, bf.v, (short)0, acc[t], false, false);
                }
            }
        }
    }

    if (!active) return;
    // Epilogue. C layout: VGPR r -> row hi*8+r, lane -> col
#pragma unroll
    for (int t = 0; t < 4; ++t) {
        const int col = col0 + t * 16 + lo;
        const float bval = bias ? bias[col] : 0.f;
#pragma unroll
        for (int r = 0; r < 8; ++r) {
            const int row = rowBase + hi * 8 + r;
            float v = acc[t][r] + bval;
            float* cp = C + (size_t)row * ldc + col;
            if (flags & 1) v += *cp;
            if (flags & 2) v = tanhf(v);
            *cp = v;
        }
    }
}

// ---------------- batched attention: logits -> masked softmax -> weighted context ----------------
// one workgroup (256 thr) per batch element
__global__ void attn_kernel(const float* __restrict__ ctx,     // [B,S,C]
                            const float* __restrict__ target,  // [B,C]
                            const int*   __restrict__ mask,    // [B,S] or null (nonzero -> -inf)
                            float* __restrict__ wctx,          // [B,C]
                            int S, int C) {
    __shared__ float slog[128];
    const int b = blockIdx.x;
    const float* cb = ctx + (size_t)b * S * C;
    const float* tb = target + (size_t)b * C;
    const int lane = threadIdx.x & 31;
    const int wave = threadIdx.x >> 5;
    const int nw = blockDim.x >> 5;

    for (int s = wave; s < S; s += nw) {
        float part = 0.f;
        const float* row = cb + (size_t)s * C;
        for (int c = lane; c < C; c += 32) part += row[c] * tb[c];
#pragma unroll
        for (int off = 16; off > 0; off >>= 1) part += __shfl_down(part, off, 32);
        if (lane == 0) {
            if (mask && mask[b * S + s]) part = -__builtin_inff();
            slog[s] = part;
        }
    }
    __syncthreads();
    if (threadIdx.x == 0) {
        float m = -__builtin_inff();
        for (int s = 0; s < S; ++s) m = fmaxf(m, slog[s]);
        float sum = 0.f;
        for (int s = 0; s < S; ++s) { float e = __expf(slog[s] - m); slog[s] = e; sum += e; }
        float inv = 1.f / sum;
        for (int s = 0; s < S; ++s) slog[s] *= inv;
    }
    __syncthreads();
    for (int c = threadIdx.x; c < C; c += blockDim.x) {
        float acc = 0.f;
        for (int s = 0; s < S; ++s) acc += slog[s] * cb[(size_t)s * C + c];
        wctx[(size_t)b * C + c] = acc;
    }
}

// ---------------- LSTM elementwise: gates[B,4H] (i,f,g,o) + c_prev -> h1, c1 ----------------
__device__ __forceinline__ float sigm(float x) { return 1.f / (1.f + __expf(-x)); }

__global__ void lstm_elem_kernel(const float* __restrict__ gates,
                                 const float* __restrict__ cprev,
                                 float* __restrict__ h1, float* __restrict__ c1, int n) {
    int i = blockIdx.x * blockDim.x + threadIdx.x;
    int stride = gridDim.x * blockDim.x;
    for (; i < n; i += stride) {
        int b = i >> 9;           // /HID
        int u = i & 511;          // %HID
        const float* g = gates + (size_t)b * GATES;
        float gi = g[u], gf = g[HIDD + u], gg = g[2 * HIDD + u], go = g[3 * HIDD + u];
        float c = sigm(gf) * cprev[i] + sigm(gi) * tanhf(gg);
        c1[i] = c;
        h1[i] = sigm(go) * tanhf(c);
    }
}

// ---------------- candidate logits: one wave per (b, nc) dot of length C ----------------
__global__ void cand_logit_kernel(const float* __restrict__ cand,    // [B,NC,C]
                                  const float* __restrict__ target,  // [B,C]
                                  float* __restrict__ out,           // [B,NC]
                                  int nc, int C, int total) {
    const int gw = (blockIdx.x * blockDim.x + threadIdx.x) >> 5;
    const int lane = threadIdx.x & 31;
    if (gw >= total) return;
    const int b = gw / nc, n = gw % nc;
    const float* row = cand + ((size_t)b * nc + n) * C;
    const float* tb = target + (size_t)b * C;
    float part = 0.f;
    for (int c = lane; c < C; c += 32) part += row[c] * tb[c];
#pragma unroll
    for (int off = 16; off > 0; off >>= 1) part += __shfl_down(part, off, 32);
    if (lane == 0) out[(size_t)b * nc + n] = part;
}

// ---------------- host side ----------------
static inline void launch_cvt(const float* s, bf16_t* d, int n, hipStream_t st) {
    int blocks = (n + 255) / 256;
    if (blocks > 4096) blocks = 4096;
    cvt_bf16_kernel<<<blocks, 256, 0, st>>>(s, d, n);
}

static inline void launch_gemm(const bf16_t* A, int lda, const bf16_t* B, int ldb,
                               float* C, int ldc, const float* bias,
                               int M, int N, int K, int flags, hipStream_t st) {
    dim3 grid((N + 511) / 512, M / 16);
    gemm_bf16_wmma<<<grid, 256, 0, st>>>(A, lda, B, ldb, C, ldc, bias, M, N, K, flags);
}

extern "C" void kernel_launch(void* const* d_in, const int* in_sizes, int n_in,
                              void* d_out, int out_size, void* d_ws, size_t ws_size,
                              hipStream_t stream) {
    // ---- inputs (setup_inputs order) ----
    const float* action    = (const float*)d_in[0];
    const float* cand_feat = (const float*)d_in[1];
    const float* prev_h1_v = (const float*)d_in[2];
    const float* prev_h1_o = (const float*)d_in[3];
    // d_in[4] = c_0_v (unused by reference, faithfully)
    const float* c_0_o     = (const float*)d_in[5];
    const float* ctx       = (const float*)d_in[6];
    const int*   ctx_mask  = (const int*)d_in[7];
    const float* feature   = (const float*)d_in[8];
    const float* denseObj  = (const float*)d_in[9];
    const int*   obj_mask  = (const int*)d_in[10];
    const float* W_emb     = (const float*)d_in[11];
    const float* b_emb     = (const float*)d_in[12];
    const float* Wih_v     = (const float*)d_in[13];
    const float* Whh_v     = (const float*)d_in[14];
    const float* bih_v     = (const float*)d_in[15];
    const float* bhh_v     = (const float*)d_in[16];
    const float* Wih_o     = (const float*)d_in[17];
    const float* Whh_o     = (const float*)d_in[18];
    const float* bih_o     = (const float*)d_in[19];
    const float* bhh_o     = (const float*)d_in[20];
    const float* Wq_feat   = (const float*)d_in[21];
    const float* Wq_dense  = (const float*)d_in[22];
    const float* Wq_av     = (const float*)d_in[23];
    const float* Wo_av     = (const float*)d_in[24];
    const float* Wq_ao     = (const float*)d_in[25];
    const float* Wo_ao     = (const float*)d_in[26];
    const float* Wq_cand   = (const float*)d_in[27];

    // ---- output slots (fp32, concatenated in return order) ----
    float* out = (float*)d_out;
    float* o_h1v  = out;                        // [512,512]
    float* o_h1o  = out + 262144;
    float* o_c1v  = out + 524288;
    float* o_c1o  = out + 786432;
    float* o_log  = out + 1048576;              // [512,16]
    float* o_htv  = out + 1056768;              // [512,512]
    float* o_hto  = out + 1318912;

    // ---- workspace bump allocator ----
    char* wsp = (char*)d_ws;
    size_t off = 0;
    auto alloc = [&](size_t bytes) -> void* {
        void* p = wsp + off;
        off = (off + bytes + 255) & ~(size_t)255;
        return p;
    };
    // bf16 weight/activation mirrors
    bf16_t* bH1vP  = (bf16_t*)alloc((size_t)BATCH * HIDD * 2);
    bf16_t* bH1oP  = (bf16_t*)alloc((size_t)BATCH * HIDD * 2);
    bf16_t* bAct   = (bf16_t*)alloc((size_t)BATCH * ANGLE * 2);
    bf16_t* bWemb  = (bf16_t*)alloc((size_t)EMBD * ANGLE * 2);
    bf16_t* bWqD   = (bf16_t*)alloc((size_t)FEATD * HIDD * 2);
    bf16_t* bWqF   = (bf16_t*)alloc((size_t)FEATD * HIDD * 2);
    bf16_t* bWihV  = (bf16_t*)alloc((size_t)GATES * 2240 * 2);
    bf16_t* bWhhV  = (bf16_t*)alloc((size_t)GATES * HIDD * 2);
    bf16_t* bWihO  = (bf16_t*)alloc((size_t)GATES * 2240 * 2);
    bf16_t* bWhhO  = (bf16_t*)alloc((size_t)GATES * HIDD * 2);
    bf16_t* bWqAv  = (bf16_t*)alloc((size_t)HIDD * HIDD * 2);
    bf16_t* bWqAo  = (bf16_t*)alloc((size_t)HIDD * HIDD * 2);
    bf16_t* bWoAv  = (bf16_t*)alloc((size_t)HIDD * 2 * HIDD * 2);
    bf16_t* bWoAo  = (bf16_t*)alloc((size_t)HIDD * 2 * HIDD * 2);
    bf16_t* bWqC   = (bf16_t*)alloc((size_t)FEATD * 2 * HIDD * 2);
    bf16_t* bEmb   = (bf16_t*)alloc((size_t)BATCH * EMBD * 2);
    bf16_t* bRN    = (bf16_t*)alloc((size_t)BATCH * FEATD * 2);
    bf16_t* bDense = (bf16_t*)alloc((size_t)BATCH * FEATD * 2);
    bf16_t* bH1v   = (bf16_t*)alloc((size_t)BATCH * HIDD * 2);
    bf16_t* bH1o   = (bf16_t*)alloc((size_t)BATCH * HIDD * 2);
    bf16_t* bWcV   = (bf16_t*)alloc((size_t)BATCH * HIDD * 2);
    bf16_t* bWcO   = (bf16_t*)alloc((size_t)BATCH * HIDD * 2);
    bf16_t* bHtV   = (bf16_t*)alloc((size_t)BATCH * HIDD * 2);
    bf16_t* bHtO   = (bf16_t*)alloc((size_t)BATCH * HIDD * 2);
    // fp32 intermediates
    float* tO      = (float*)alloc((size_t)BATCH * FEATD * 4);
    float* tV      = (float*)alloc((size_t)BATCH * FEATD * 4);
    float* embF    = (float*)alloc((size_t)BATCH * EMBD * 4);
    float* rnF     = (float*)alloc((size_t)BATCH * FEATD * 4);
    float* denseF  = (float*)alloc((size_t)BATCH * FEATD * 4);
    float* gV      = (float*)alloc((size_t)BATCH * GATES * 4);
    float* gO      = (float*)alloc((size_t)BATCH * GATES * 4);
    float* tAv     = (float*)alloc((size_t)BATCH * HIDD * 4);
    float* tAo     = (float*)alloc((size_t)BATCH * HIDD * 4);
    float* wcVf    = (float*)alloc((size_t)BATCH * HIDD * 4);
    float* wcOf    = (float*)alloc((size_t)BATCH * HIDD * 4);
    float* tCand   = (float*)alloc((size_t)BATCH * FEATD * 4);
    (void)ws_size; (void)in_sizes; (void)n_in; (void)out_size;

    // ---- 1: convert weights + initial activations to bf16 ----
    launch_cvt(prev_h1_v, bH1vP, BATCH * HIDD, stream);
    launch_cvt(prev_h1_o, bH1oP, BATCH * HIDD, stream);
    launch_cvt(action,    bAct,  BATCH * ANGLE, stream);
    launch_cvt(W_emb,     bWemb, EMBD * ANGLE, stream);
    launch_cvt(Wq_dense,  bWqD,  FEATD * HIDD, stream);
    launch_cvt(Wq_feat,   bWqF,  FEATD * HIDD, stream);
    launch_cvt(Wih_v,     bWihV, GATES * 2240, stream);
    launch_cvt(Whh_v,     bWhhV, GATES * HIDD, stream);
    launch_cvt(Wih_o,     bWihO, GATES * 2240, stream);
    launch_cvt(Whh_o,     bWhhO, GATES * HIDD, stream);
    launch_cvt(Wq_av,     bWqAv, HIDD * HIDD, stream);
    launch_cvt(Wq_ao,     bWqAo, HIDD * HIDD, stream);
    launch_cvt(Wo_av,     bWoAv, HIDD * 2 * HIDD, stream);
    launch_cvt(Wo_ao,     bWoAo, HIDD * 2 * HIDD, stream);
    launch_cvt(Wq_cand,   bWqC,  FEATD * 2 * HIDD, stream);

    // ---- 2: attention query projections (WMMA) ----
    launch_gemm(bH1oP, HIDD, bWqD, HIDD, tO, FEATD, nullptr, BATCH, FEATD, HIDD, 0, stream);
    launch_gemm(bH1vP, HIDD, bWqF, HIDD, tV, FEATD, nullptr, BATCH, FEATD, HIDD, 0, stream);
    // action embedding: tanh(action @ W_emb^T + b_emb)
    launch_gemm(bAct, ANGLE, bWemb, ANGLE, embF, EMBD, b_emb, BATCH, EMBD, ANGLE, 2, stream);

    // ---- 3: feature / denseObj attention (bandwidth-bound GEMV+softmax) ----
    attn_kernel<<<BATCH, 256, 0, stream>>>(denseObj, tO, obj_mask, denseF, NV, FEATD);
    attn_kernel<<<BATCH, 256, 0, stream>>>(feature,  tV, nullptr,  rnF,   NV, FEATD);

    launch_cvt(embF,   bEmb,   BATCH * EMBD, stream);
    launch_cvt(rnF,    bRN,    BATCH * FEATD, stream);
    launch_cvt(denseF, bDense, BATCH * FEATD, stream);

    // ---- 4: LSTM gates (concat folded into split-K GEMMs) ----
    // gates_v = emb @ Wih_v[:, :64]^T + rn @ Wih_v[:, 64:]^T + h_prev @ Whh_v^T + bih + bhh
    launch_gemm(bEmb,  EMBD,  bWihV,      2240, gV, GATES, bih_v, BATCH, GATES, EMBD,  0, stream);
    launch_gemm(bRN,   FEATD, bWihV + 64, 2240, gV, GATES, bhh_v, BATCH, GATES, FEATD, 1, stream);
    launch_gemm(bH1vP, HIDD,  bWhhV,      HIDD, gV, GATES, nullptr, BATCH, GATES, HIDD, 1, stream);
    launch_gemm(bEmb,  EMBD,  bWihO,      2240, gO, GATES, bih_o, BATCH, GATES, EMBD,  0, stream);
    launch_gemm(bDense,FEATD, bWihO + 64, 2240, gO, GATES, bhh_o, BATCH, GATES, FEATD, 1, stream);
    launch_gemm(bH1oP, HIDD,  bWhhO,      HIDD, gO, GATES, nullptr, BATCH, GATES, HIDD, 1, stream);

    // ---- 5: LSTM elementwise (NOTE: reference uses c_0_o for BOTH cells) ----
    lstm_elem_kernel<<<(BATCH * HIDD + 255) / 256, 256, 0, stream>>>(gV, c_0_o, o_h1v, o_c1v, BATCH * HIDD);
    lstm_elem_kernel<<<(BATCH * HIDD + 255) / 256, 256, 0, stream>>>(gO, c_0_o, o_h1o, o_c1o, BATCH * HIDD);

    launch_cvt(o_h1v, bH1v, BATCH * HIDD, stream);
    launch_cvt(o_h1o, bH1o, BATCH * HIDD, stream);

    // ---- 6: ctx attention for both streams ----
    launch_gemm(bH1v, HIDD, bWqAv, HIDD, tAv, HIDD, nullptr, BATCH, HIDD, HIDD, 0, stream);
    launch_gemm(bH1o, HIDD, bWqAo, HIDD, tAo, HIDD, nullptr, BATCH, HIDD, HIDD, 0, stream);
    attn_kernel<<<BATCH, 256, 0, stream>>>(ctx, tAv, ctx_mask, wcVf, SEQ, HIDD);
    attn_kernel<<<BATCH, 256, 0, stream>>>(ctx, tAo, ctx_mask, wcOf, SEQ, HIDD);

    launch_cvt(wcVf, bWcV, BATCH * HIDD, stream);
    launch_cvt(wcOf, bWcO, BATCH * HIDD, stream);

    // ---- 7: h_tilde = tanh([wctx, h] @ Wo^T) (split-K, tanh on final accumulate) ----
    launch_gemm(bWcV, HIDD, bWoAv,        2 * HIDD, o_htv, HIDD, nullptr, BATCH, HIDD, HIDD, 0, stream);
    launch_gemm(bH1v, HIDD, bWoAv + HIDD, 2 * HIDD, o_htv, HIDD, nullptr, BATCH, HIDD, HIDD, 3, stream);
    launch_gemm(bWcO, HIDD, bWoAo,        2 * HIDD, o_hto, HIDD, nullptr, BATCH, HIDD, HIDD, 0, stream);
    launch_gemm(bH1o, HIDD, bWoAo + HIDD, 2 * HIDD, o_hto, HIDD, nullptr, BATCH, HIDD, HIDD, 3, stream);

    launch_cvt(o_htv, bHtV, BATCH * HIDD, stream);
    launch_cvt(o_hto, bHtO, BATCH * HIDD, stream);

    // ---- 8: candidate attention logits ----
    launch_gemm(bHtV, HIDD, bWqC,        2 * HIDD, tCand, FEATD, nullptr, BATCH, FEATD, HIDD, 0, stream);
    launch_gemm(bHtO, HIDD, bWqC + HIDD, 2 * HIDD, tCand, FEATD, nullptr, BATCH, FEATD, HIDD, 1, stream);
    cand_logit_kernel<<<(BATCH * NCAND) / 8, 256, 0, stream>>>(cand_feat, tCand, o_log, NCAND, FEATD, BATCH * NCAND);
}
